// BinaryLinearLayer_7035156431032
// MI455X (gfx1250) — compile-verified, matrix operand order checked by reading, not run on hardware
//
#include <hip/hip_runtime.h>

// y[8192,4096] = x[8192,4096] @ sign(w[4096,4096])^T + sign(b[4096])
#define BATCH 8192
#define KDIM  4096
#define NDIM  4096

// bf16 GEMM (workspace path): block tile 256x128, K-step 64, 8 waves, wave tile 64x64
#define BMG 256
#define BNG 128
#define BKB 64
// fused fallback: block tile 128x128, fp32 staged, K-step 32
#define BM 128
#define BN 128
#define BKF 32

typedef __attribute__((ext_vector_type(16))) __bf16 v16bf;
typedef __attribute__((ext_vector_type(8)))  __bf16 v8bf;
typedef __attribute__((ext_vector_type(8)))  float  v8f;
typedef __attribute__((ext_vector_type(4)))  float  v4f;

#if defined(__HIP_DEVICE_COMPILE__) && \
    __has_builtin(__builtin_amdgcn_global_load_async_to_lds_b128) && \
    __has_builtin(__builtin_amdgcn_s_wait_asynccnt)
#define USE_ASYNC 1
// builtin prototype (probe-confirmed): (v4i AS1*, v4i AS3*, imm int, imm int)
typedef int v4i_b __attribute__((vector_size(4 * sizeof(int))));
typedef __attribute__((address_space(1))) v4i_b* gv4i_p;
typedef __attribute__((address_space(3))) v4i_b* lv4i_p;
#else
#define USE_ASYNC 0
#endif

__device__ __forceinline__ float fsignf(float v) {
    return (v > 0.0f) ? 1.0f : ((v < 0.0f) ? -1.0f : 0.0f);
}

// sign() directly as a bf16 bit pattern: {-1,0,+1} -> {0xBF80, 0x0000, 0x3F80}
__device__ __forceinline__ __bf16 bsign(float v) {
    unsigned u = __builtin_bit_cast(unsigned, v);
    unsigned s = ((u & 0x7fffffffu) == 0u) ? 0u : (0x3f80u | ((u >> 16) & 0x8000u));
    return __builtin_bit_cast(__bf16, (unsigned short)s);
}

// 16x32 bf16 A-fragment (B identical since w is K-major) from a K-major LDS tile
// [rows][STRIDE].  CDNA5 ISA 7.12.2 layout:
//   lanes 0-15 : row = lane,    VGPR0-3 = K 0..7,  VGPR4-7 = K 16..23
//   lanes16-31 : row = lane-16, VGPR0-3 = K 8..15, VGPR4-7 = K 24..31
template <int STRIDE>
__device__ __forceinline__ v16bf ld_frag(const __bf16* tile, int rowbase, int kk, int lane) {
    const int row = rowbase + (lane & 15);
    const int kh  = (lane >> 4) * 8;
    const __bf16* p = tile + row * STRIDE + kk + kh;
    v8bf lo = *(const v8bf*)(p);
    v8bf hi = *(const v8bf*)(p + 16);
    return __builtin_shufflevector(lo, hi, 0,1,2,3,4,5,6,7,8,9,10,11,12,13,14,15);
}

// One K=32 slab: MT x NT WMMA tiles for this wave.
template <int STRIDE, int MT, int NT>
__device__ __forceinline__ void wmma_step(const __bf16* sx, const __bf16* sw, int kk,
                                          v8f (&acc)[MT][NT], int baseM, int baseN, int lane) {
    v16bf a[MT], b[NT];
    #pragma unroll
    for (int mi = 0; mi < MT; ++mi) a[mi] = ld_frag<STRIDE>(sx, baseM + mi * 16, kk, lane);
    #pragma unroll
    for (int ni = 0; ni < NT; ++ni) b[ni] = ld_frag<STRIDE>(sw, baseN + ni * 16, kk, lane);
    #pragma unroll
    for (int mi = 0; mi < MT; ++mi)
        #pragma unroll
        for (int ni = 0; ni < NT; ++ni)
            acc[mi][ni] = __builtin_amdgcn_wmma_f32_16x16x32_bf16(
                false, a[mi], false, b[ni], (short)0, acc[mi][ni], false, false);
}

// C/D layout: VGPR r -> M = r + (lane>=16)*8, N = lane&15.  NT stores (write-once output).
template <int MT, int NT>
__device__ __forceinline__ void epilogue(v8f (&acc)[MT][NT], const float* bias, float* out,
                                         int gM, int gN, int baseM, int baseN, int lane) {
    const int col  = lane & 15;
    const int rsel = (lane >> 4) * 8;
    #pragma unroll
    for (int ni = 0; ni < NT; ++ni) {
        const int ncol = gN + baseN + ni * 16 + col;
        const float bs = fsignf(bias[ncol]);
        #pragma unroll
        for (int mi = 0; mi < MT; ++mi) {
            #pragma unroll
            for (int r = 0; r < 8; ++r) {
                const int row = gM + baseM + mi * 16 + rsel + r;
                __builtin_nontemporal_store(acc[mi][ni][r] + bs,
                                            &out[(size_t)row * NDIM + ncol]);
            }
        }
    }
}

// ---------------- pre-pass kernels (workspace path) ----------------

__global__ __launch_bounds__(256)
void prep_x_bf16(const float* __restrict__ x, __bf16* __restrict__ xb) {
    const size_t i = ((size_t)blockIdx.x * 256 + threadIdx.x) * 8;
    v4f a = __builtin_nontemporal_load((const v4f*)(x + i));
    v4f b = __builtin_nontemporal_load((const v4f*)(x + i + 4));
    v8bf o;
    #pragma unroll
    for (int j = 0; j < 4; ++j) { o[j] = (__bf16)a[j]; o[4 + j] = (__bf16)b[j]; }
    *(v8bf*)(xb + i) = o;   // regular store: stays resident in L2 for the GEMM
}

__global__ __launch_bounds__(256)
void prep_w_sign(const float* __restrict__ w, __bf16* __restrict__ wb) {
    const size_t i = ((size_t)blockIdx.x * 256 + threadIdx.x) * 8;
    v4f a = __builtin_nontemporal_load((const v4f*)(w + i));
    v4f b = __builtin_nontemporal_load((const v4f*)(w + i + 4));
    v8bf o;
    #pragma unroll
    for (int j = 0; j < 4; ++j) { o[j] = bsign(a[j]); o[4 + j] = bsign(b[j]); }
    *(v8bf*)(wb + i) = o;
}

// ---------------- main GEMM, bf16 operands, 256x128 block / 64x64 wave ----------------

__global__ __launch_bounds__(256)
void gemm_wmma_bf16(const __bf16* __restrict__ xb, const __bf16* __restrict__ wb,
                    const float* __restrict__ bias, float* __restrict__ out) {
    __shared__ __bf16 sx[2][BMG * BKB];   // 2 x 32 KB
    __shared__ __bf16 sw[2][BNG * BKB];   // 2 x 16 KB

    const int tid   = threadIdx.x;
    const int lane  = tid & 31;
    const int wid   = tid >> 5;           // 8 waves: 4 (M) x 2 (N)
    const int baseM = (wid >> 1) * 64;
    const int baseN = (wid & 1) * 64;

    const int gM = blockIdx.y * BMG;
    const int gN = blockIdx.x * BNG;

    // x loader: 1 thread per row, 64 contiguous bf16 (128 B)
    // w loader: 2 threads per row, 32 contiguous bf16 (64 B)
    const __bf16* xg = xb + (size_t)(gM + tid) * KDIM;
    const __bf16* wg = wb + (size_t)(gN + (tid >> 1)) * KDIM + (tid & 1) * 32;
    __bf16* xd0 = &sx[0][tid * BKB];
    __bf16* wd0 = &sw[0][(tid >> 1) * BKB + (tid & 1) * 32];

    v8f acc[4][4];
    #pragma unroll
    for (int mi = 0; mi < 4; ++mi)
        #pragma unroll
        for (int ni = 0; ni < 4; ++ni)
            #pragma unroll
            for (int e = 0; e < 8; ++e) acc[mi][ni][e] = 0.0f;

#if USE_ASYNC
    // gfx1250 async global->LDS staging (ASYNCcnt): 12 x b128 per stage per thread
    auto issue = [&](int buf, int k0) {
        const __bf16* xs = xg + k0;
        const __bf16* ws = wg + k0;
        __bf16* xd = xd0 + buf * (BMG * BKB);
        __bf16* wd = wd0 + buf * (BNG * BKB);
        #pragma unroll
        for (int j = 0; j < 8; ++j)
            __builtin_amdgcn_global_load_async_to_lds_b128(
                (gv4i_p)(xs + 8 * j), (lv4i_p)(xd + 8 * j), 0, 0);
        #pragma unroll
        for (int j = 0; j < 4; ++j)
            __builtin_amdgcn_global_load_async_to_lds_b128(
                (gv4i_p)(ws + 8 * j), (lv4i_p)(wd + 8 * j), 0, 0);
    };
    issue(0, 0);
    int cur = 0;
    for (int k0 = BKB; k0 < KDIM; k0 += BKB) {
        issue(cur ^ 1, k0);                      // overlap next stage with compute
        __builtin_amdgcn_s_wait_asynccnt(12);    // previous stage's 12 asyncs done
        __syncthreads();
        #pragma unroll
        for (int kk = 0; kk < BKB; kk += 32)
            wmma_step<BKB>(sx[cur], sw[cur], kk, acc, baseM, baseN, lane);
        __syncthreads();                         // everyone done reading buf before reuse
        cur ^= 1;
    }
    __builtin_amdgcn_s_wait_asynccnt(0);
    __syncthreads();
    #pragma unroll
    for (int kk = 0; kk < BKB; kk += 32)
        wmma_step<BKB>(sx[cur], sw[cur], kk, acc, baseM, baseN, lane);
#else
    // sync staging, register double buffer
    {
        uint4 xr[8], wr[4];
        #pragma unroll
        for (int j = 0; j < 8; ++j) xr[j] = *(const uint4*)(xg + 8 * j);
        #pragma unroll
        for (int j = 0; j < 4; ++j) wr[j] = *(const uint4*)(wg + 8 * j);
        #pragma unroll
        for (int j = 0; j < 8; ++j) *(uint4*)(xd0 + 8 * j) = xr[j];
        #pragma unroll
        for (int j = 0; j < 4; ++j) *(uint4*)(wd0 + 8 * j) = wr[j];
    }
    __syncthreads();
    int cur = 0;
    for (int k0 = BKB; k0 < KDIM; k0 += BKB) {
        uint4 xr[8], wr[4];
        #pragma unroll
        for (int j = 0; j < 8; ++j) xr[j] = *(const uint4*)(xg + k0 + 8 * j);
        #pragma unroll
        for (int j = 0; j < 4; ++j) wr[j] = *(const uint4*)(wg + k0 + 8 * j);
        #pragma unroll
        for (int kk = 0; kk < BKB; kk += 32)
            wmma_step<BKB>(sx[cur], sw[cur], kk, acc, baseM, baseN, lane);
        __bf16* xd = xd0 + (cur ^ 1) * (BMG * BKB);
        __bf16* wd = wd0 + (cur ^ 1) * (BNG * BKB);
        #pragma unroll
        for (int j = 0; j < 8; ++j) *(uint4*)(xd + 8 * j) = xr[j];
        #pragma unroll
        for (int j = 0; j < 4; ++j) *(uint4*)(wd + 8 * j) = wr[j];
        __syncthreads();
        cur ^= 1;
    }
    #pragma unroll
    for (int kk = 0; kk < BKB; kk += 32)
        wmma_step<BKB>(sx[cur], sw[cur], kk, acc, baseM, baseN, lane);
#endif

    epilogue(acc, bias, out, gM, gN, baseM, baseN, lane);
}

// ---------------- fused fallback (workspace too small): fp32 staged + inline convert ----

__device__ __forceinline__ void cvt_store(__bf16* dx, __bf16* dw,
                                          const float4* xv, const float4* wv,
                                          int lrow, int lcol) {
    v8bf xb0, xb1, wb0, wb1;
    #pragma unroll
    for (int i = 0; i < 4; ++i) {
        const float* xf = (const float*)&xv[i];
        const float* wf = (const float*)&wv[i];
        #pragma unroll
        for (int j = 0; j < 4; ++j) {
            const int idx = 4 * i + j;
            __bf16 xc = (__bf16)xf[j];
            __bf16 wc = bsign(wf[j]);
            if (idx < 8) { xb0[idx] = xc;     wb0[idx] = wc;     }
            else         { xb1[idx - 8] = xc; wb1[idx - 8] = wc; }
        }
    }
    *(v8bf*)(dx + lrow * BKF + lcol)     = xb0;
    *(v8bf*)(dx + lrow * BKF + lcol + 8) = xb1;
    *(v8bf*)(dw + lrow * BKF + lcol)     = wb0;
    *(v8bf*)(dw + lrow * BKF + lcol + 8) = wb1;
}

__global__ __launch_bounds__(256)
void binlin_fused_kernel(const float* __restrict__ x, const float* __restrict__ w,
                         const float* __restrict__ bias, float* __restrict__ out) {
    __shared__ __bf16 sx[2][BM * BKF];
    __shared__ __bf16 sw[2][BN * BKF];

    const int tid   = threadIdx.x;
    const int lane  = tid & 31;
    const int wid   = tid >> 5;
    const int baseM = (wid >> 2) * 64;   // 2 waves in M
    const int baseN = (wid & 3) * 32;    // 4 waves in N

    const int gM = blockIdx.y * BM;
    const int gN = blockIdx.x * BN;

    const int lrow = tid >> 1;
    const int lcol = (tid & 1) * 16;

    const float* xptr = x + (size_t)(gM + lrow) * KDIM + lcol;
    const float* wptr = w + (size_t)(gN + lrow) * KDIM + lcol;

    v8f acc[4][2];
    #pragma unroll
    for (int mi = 0; mi < 4; ++mi)
        #pragma unroll
        for (int ni = 0; ni < 2; ++ni)
            #pragma unroll
            for (int e = 0; e < 8; ++e) acc[mi][ni][e] = 0.0f;

    {
        float4 xv[4], wv[4];
        #pragma unroll
        for (int i = 0; i < 4; ++i) {
            xv[i] = *(const float4*)(xptr + 4 * i);
            wv[i] = *(const float4*)(wptr + 4 * i);
        }
        cvt_store(sx[0], sw[0], xv, wv, lrow, lcol);
    }
    __syncthreads();

    int cur = 0;
    for (int k0 = BKF; k0 < KDIM; k0 += BKF) {
        float4 xv[4], wv[4];
        #pragma unroll
        for (int i = 0; i < 4; ++i) {
            xv[i] = *(const float4*)(xptr + k0 + 4 * i);
            wv[i] = *(const float4*)(wptr + k0 + 4 * i);
        }
        if (k0 + BKF < KDIM) {
            __builtin_prefetch(xptr + k0 + BKF, 0, 1);
            __builtin_prefetch(wptr + k0 + BKF, 0, 1);
        }
        wmma_step<BKF>(sx[cur], sw[cur], 0, acc, baseM, baseN, lane);
        cvt_store(sx[cur ^ 1], sw[cur ^ 1], xv, wv, lrow, lcol);
        __syncthreads();
        cur ^= 1;
    }
    wmma_step<BKF>(sx[cur], sw[cur], 0, acc, baseM, baseN, lane);

    epilogue(acc, bias, out, gM, gN, baseM, baseN, lane);
}

// ---------------- launch ----------------

extern "C" void kernel_launch(void* const* d_in, const int* in_sizes, int n_in,
                              void* d_out, int out_size, void* d_ws, size_t ws_size,
                              hipStream_t stream) {
    const float* x    = (const float*)d_in[0];   // [8192, 4096] fp32
    const float* w    = (const float*)d_in[1];   // [4096, 4096] fp32
    const float* bias = (const float*)d_in[2];   // [4096] fp32
    float* out        = (float*)d_out;           // [8192, 4096] fp32

    const size_t need = ((size_t)BATCH * KDIM + (size_t)NDIM * KDIM) * sizeof(__bf16);

    if (ws_size >= need) {
        __bf16* xb = (__bf16*)d_ws;
        __bf16* wb = xb + (size_t)BATCH * KDIM;
        prep_x_bf16<<<(BATCH * (size_t)KDIM) / (256 * 8), 256, 0, stream>>>(x, xb);
        prep_w_sign<<<(NDIM * (size_t)KDIM) / (256 * 8), 256, 0, stream>>>(w, wb);
        dim3 grid(NDIM / BNG, BATCH / BMG);      // 32 x 32 workgroups
        gemm_wmma_bf16<<<grid, 256, 0, stream>>>(xb, wb, bias, out);
    } else {
        dim3 grid(NDIM / BN, BATCH / BM);        // 32 x 64 workgroups
        binlin_fused_kernel<<<grid, 256, 0, stream>>>(x, w, bias, out);
    }
}
